// attention_69337952027165
// MI455X (gfx1250) — compile-verified, matrix operand order checked by reading, not run on hardware
//
#include <hip/hip_runtime.h>

#define CTX  2048
#define DIMS 1024
#define HEAD 16
#define HD   64

typedef __attribute__((ext_vector_type(16))) _Float16 v16h;
typedef __attribute__((ext_vector_type(8)))  _Float16 v8h;
typedef __attribute__((ext_vector_type(2)))  _Float16 v2h;
typedef __attribute__((ext_vector_type(8)))  float    v8f;

// Build a 16x16x32 f16 WMMA A/B fragment for this lane.
// 16-bit operand layout (ISA 7.12.2): lane half (lane/16) selects
// K in {0..7,16..23} (half=0) or {8..15,24..31} (half=1); both chunks are
// contiguous 8 x f16 = 16B runs, so two aligned 16-byte loads suffice.
__device__ inline v16h load_frag(const _Float16* rowbase, int half) {
  v8h lo = *(const v8h*)(rowbase + half * 8);
  v8h hi = *(const v8h*)(rowbase + 16 + half * 8);
  v16h r;
#pragma unroll
  for (int i = 0; i < 8; ++i) { r[i] = lo[i]; r[i + 8] = hi[i]; }
  return r;
}

// ---------------------------------------------------------------------------
// Stage 1: fused LayerNorm (q-path & kv-path share mean/var) + rotary radius
// ---------------------------------------------------------------------------
__global__ __launch_bounds__(256) void prep_kernel(
    const float* __restrict__ x,
    const float* __restrict__ qn_w, const float* __restrict__ qn_b,
    const float* __restrict__ kvn_w, const float* __restrict__ kvn_b,
    const float* __restrict__ rh_w, const float* __restrict__ rh_b,
    _Float16* __restrict__ lnq, _Float16* __restrict__ lnkv,
    float* __restrict__ radius)
{
  const int c = blockIdx.x, t = threadIdx.x;
  const float* row = x + (size_t)c * DIMS;
  float v[4];
  float s = 0.f, s2 = 0.f, dp = 0.f;
#pragma unroll
  for (int i = 0; i < 4; ++i) {
    int d = t + i * 256;
    float val = row[d];
    v[i] = val;
    s += val; s2 += val * val; dp += val * rh_w[d];
  }
  __shared__ float red[256];
  auto reduceAll = [&](float val) -> float {
    red[t] = val; __syncthreads();
    for (int o = 128; o > 0; o >>= 1) { if (t < o) red[t] += red[t + o]; __syncthreads(); }
    float r = red[0]; __syncthreads(); return r;
  };
  float tot  = reduceAll(s);
  float tot2 = reduceAll(s2);
  float dtot = reduceAll(dp);
  float mean = tot * (1.f / DIMS);
  float var  = tot2 * (1.f / DIMS) - mean * mean;
  float inv  = rsqrtf(var + 1e-5f);
  if (t == 0) {
    float per = dtot + rh_b[0];
    radius[c] = fminf(fmaxf(per * 0.01f, 0.5f), 2.0f);
  }
#pragma unroll
  for (int i = 0; i < 4; ++i) {
    int d = t + i * 256;
    float n = (v[i] - mean) * inv;
    lnq [(size_t)c * DIMS + d] = (_Float16)(n * qn_w[d] + qn_b[d]);
    lnkv[(size_t)c * DIMS + d] = (_Float16)(n * kvn_w[d] + kvn_b[d]);
  }
}

// ---------------------------------------------------------------------------
// Stage 2: f32 -> f16 weight conversion (one pass; weights then live in L2)
// ---------------------------------------------------------------------------
__global__ void cvt_kernel(const float* __restrict__ s, _Float16* __restrict__ d, int n) {
  int i = blockIdx.x * blockDim.x + threadIdx.x;
  int stride = gridDim.x * blockDim.x;
  for (; i < n; i += stride) d[i] = (_Float16)s[i];
}

// ---------------------------------------------------------------------------
// Stage 3/6: LDS-staged WMMA GEMM.  C(f32) = A(f16,MxK) * B(f16,KxN) + bias
// Block tile 64x128x32; 8 waves in 4(M) x 2(N); each wave owns a 16x64 strip
// = 4 accumulators -> 4 WMMAs per K-step sharing one A-fragment.
// LDS rows padded to 48 f16 (96 B) so all 16-byte fragment chunks stay aligned.
// ---------------------------------------------------------------------------
#define BM 64
#define BN 128
#define BK 32
#define LP 48   // padded LDS row stride in f16 (96 bytes, 16B-aligned chunks)
__global__ __launch_bounds__(256) void gemm_kernel(
    const _Float16* __restrict__ A, const _Float16* __restrict__ B,
    const float* __restrict__ bias, float* __restrict__ C,
    int M, int N, int K)
{
  __shared__ __align__(16) _Float16 As[BM][LP];
  __shared__ __align__(16) _Float16 Bt[BN][LP];   // B staged transposed (NxK)
  const int t = threadIdx.x, lane = t & 31, w = t >> 5;
  const int wm = w >> 1, wn = w & 1;              // 4 x 2 wave grid
  const int half = lane >> 4, l16 = lane & 15;
  const int bm = blockIdx.y * BM, bn = blockIdx.x * BN;
  const int ar = t >> 2, ac = (t & 3) * 8;        // A: one b128 per thread
  const int kp = t >> 4, ng = t & 15;             // B: two rows x 8 cols per thread
  v8f acc0 = {}, acc1 = {}, acc2 = {}, acc3 = {};
  for (int k0 = 0; k0 < K; k0 += BK) {
    // stage A tile (64x32): vectorized 16B load + 16B LDS store
    *(v8h*)&As[ar][ac] = *(const v8h*)(A + (size_t)(bm + ar) * K + k0 + ac);
    // stage B tile (32x128) transposed: pack two K-consecutive f16 per b32 store
    v8h blo = *(const v8h*)(B + (size_t)(k0 + 2 * kp)     * N + bn + ng * 8);
    v8h bhi = *(const v8h*)(B + (size_t)(k0 + 2 * kp + 1) * N + bn + ng * 8);
#pragma unroll
    for (int i = 0; i < 8; ++i) {
      v2h p; p[0] = blo[i]; p[1] = bhi[i];
      *(v2h*)&Bt[ng * 8 + i][2 * kp] = p;
    }
    if (k0 + BK < K) {  // prefetch next K tile -> global_prefetch_b8
      __builtin_prefetch(A + (size_t)(bm + ar) * K + k0 + BK + ac, 0, 3);
      __builtin_prefetch(B + (size_t)(k0 + BK + 2 * kp) * N + bn + ng * 8, 0, 3);
    }
    __syncthreads();
    const v16h a = load_frag(&As[wm * 16 + l16][0], half);
    const v16h b0 = load_frag(&Bt[wn * 64 +  0 + l16][0], half);
    const v16h b1 = load_frag(&Bt[wn * 64 + 16 + l16][0], half);
    const v16h b2 = load_frag(&Bt[wn * 64 + 32 + l16][0], half);
    const v16h b3 = load_frag(&Bt[wn * 64 + 48 + l16][0], half);
    acc0 = __builtin_amdgcn_wmma_f32_16x16x32_f16(false, a, false, b0, (short)0, acc0, false, false);
    acc1 = __builtin_amdgcn_wmma_f32_16x16x32_f16(false, a, false, b1, (short)0, acc1, false, false);
    acc2 = __builtin_amdgcn_wmma_f32_16x16x32_f16(false, a, false, b2, (short)0, acc2, false, false);
    acc3 = __builtin_amdgcn_wmma_f32_16x16x32_f16(false, a, false, b3, (short)0, acc3, false, false);
    __syncthreads();
  }
  // epilogue: C layout per tile: M = r + 8*half, N = l16
  const int ncol0 = bn + wn * 64 + l16;
  const float bb0 = bias[ncol0], bb1 = bias[ncol0 + 16];
  const float bb2 = bias[ncol0 + 32], bb3 = bias[ncol0 + 48];
#pragma unroll
  for (int r = 0; r < 8; ++r) {
    const size_t rowoff = (size_t)(bm + wm * 16 + r + 8 * half) * N + ncol0;
    C[rowoff]      = acc0[r] + bb0;
    C[rowoff + 16] = acc1[r] + bb1;
    C[rowoff + 32] = acc2[r] + bb2;
    C[rowoff + 48] = acc3[r] + bb3;
  }
}

// ---------------------------------------------------------------------------
// Stage 4: rotary (mel-scaled freqs, per-step radius) + per-head LayerNorm.
// One wave per (ctx row, head); lane j handles complex pair (2j, 2j+1).
// Also writes V transposed [h][d][c] so flash V fragments are contiguous.
// ---------------------------------------------------------------------------
__global__ __launch_bounds__(32) void rotln_kernel(
    const float* __restrict__ qf, const float* __restrict__ kvf,
    const float* __restrict__ radius,
    const float* __restrict__ ln_w, const float* __restrict__ ln_b,
    _Float16* __restrict__ qn, _Float16* __restrict__ kn,
    _Float16* __restrict__ vT)
{
  const int c = blockIdx.x, h = blockIdx.y, j = threadIdx.x;
  const float mel_max = 2595.0f * 1.3222192947339193f;          // 2595*log10(21)
  const float lin = (float)j * (mel_max / 31.0f);               // linspace, 32 pts
  const float fb = 0.2f * (expf(lin * (2.3025850929940457f / 2595.0f)) - 1.0f);
  const float freq = (float)c * fb;
  const float rad = radius[c];
  const float fr = rad * cosf(freq), fi = rad * sinf(freq);
  const int d0 = 2 * j, d1 = 2 * j + 1;

  const float* qrow = qf + (size_t)c * DIMS + h * HD;
  float qr = qrow[d0], qi = qrow[d1];
  float qor = qr * fr - qi * fi, qoi = qr * fi + qi * fr;

  const float* krow = kvf + (size_t)c * (2 * DIMS) + h * HD;
  float kr = krow[d0], ki = krow[d1];
  float kor = kr * fr - ki * fi, koi = kr * fi + ki * fr;

  const float* vrow = kvf + (size_t)c * (2 * DIMS) + DIMS + h * HD;
  vT[(size_t)(h * HD + d0) * CTX + c] = (_Float16)vrow[d0];
  vT[(size_t)(h * HD + d1) * CTX + c] = (_Float16)vrow[d1];

  // per-head LN over 64 elems, reduced across the 32 lanes of this wave
  float qs = qor + qoi, qs2 = qor * qor + qoi * qoi;
  float ks = kor + koi, ks2 = kor * kor + koi * koi;
#pragma unroll
  for (int m = 16; m >= 1; m >>= 1) {
    qs += __shfl_xor(qs, m, 32);  qs2 += __shfl_xor(qs2, m, 32);
    ks += __shfl_xor(ks, m, 32);  ks2 += __shfl_xor(ks2, m, 32);
  }
  float qm = qs * (1.f / HD), qv = qs2 * (1.f / HD) - qm * qm, qiv = rsqrtf(qv + 1e-5f);
  float km = ks * (1.f / HD), kv = ks2 * (1.f / HD) - km * km, kiv = rsqrtf(kv + 1e-5f);
  _Float16* qdst = qn + ((size_t)h * CTX + c) * HD;
  _Float16* kdst = kn + ((size_t)h * CTX + c) * HD;
  qdst[d0] = (_Float16)((qor - qm) * qiv * ln_w[d0] + ln_b[d0]);
  qdst[d1] = (_Float16)((qoi - qm) * qiv * ln_w[d1] + ln_b[d1]);
  kdst[d0] = (_Float16)((kor - km) * kiv * ln_w[d0] + ln_b[d0]);
  kdst[d1] = (_Float16)((koi - km) * kiv * ln_w[d1] + ln_b[d1]);
}

// ---------------------------------------------------------------------------
// Stage 5: flash attention. 4 waves/block; each wave owns 16 q rows of one
// head, streams 32 keys/iter: 4 WMMAs for S, online softmax (shfl reductions
// per 16-lane half matching C layout), P re-shaped via small LDS tile,
// 4 WMMAs for O += P*V. Scores never hit HBM.
// ---------------------------------------------------------------------------
__global__ __launch_bounds__(128) void flash_kernel(
    const _Float16* __restrict__ qn, const _Float16* __restrict__ kn,
    const _Float16* __restrict__ vT, _Float16* __restrict__ aout)
{
  const int lane = threadIdx.x & 31;
  const int w = threadIdx.x >> 5;
  const int h = blockIdx.y;
  const int qt = blockIdx.x * 4 + w;          // 16-row query tile
  const int half = lane >> 4, l16 = lane & 15;
  __shared__ __align__(16) _Float16 P[4][16][40];

  const _Float16* qrow = qn + ((size_t)h * CTX + qt * 16 + l16) * HD;
  const v16h aq0 = load_frag(qrow, half);       // K = 0..31
  const v16h aq1 = load_frag(qrow + 32, half);  // K = 32..63

  v8f o0 = {}, o1 = {}, o2 = {}, o3 = {};
  float mrow[8], lrow[8];
#pragma unroll
  for (int r = 0; r < 8; ++r) { mrow[r] = -1e30f; lrow[r] = 0.f; }

  for (int kb = 0; kb < CTX; kb += 32) {
    const _Float16* k0row = kn + ((size_t)h * CTX + kb + l16) * HD;
    const _Float16* k1row = k0row + 16 * HD;
    v8f z = {};
    v8f s0 = __builtin_amdgcn_wmma_f32_16x16x32_f16(false, aq0, false, load_frag(k0row, half), (short)0, z, false, false);
    s0 = __builtin_amdgcn_wmma_f32_16x16x32_f16(false, aq1, false, load_frag(k0row + 32, half), (short)0, s0, false, false);
    v8f s1 = __builtin_amdgcn_wmma_f32_16x16x32_f16(false, aq0, false, load_frag(k1row, half), (short)0, z, false, false);
    s1 = __builtin_amdgcn_wmma_f32_16x16x32_f16(false, aq1, false, load_frag(k1row + 32, half), (short)0, s1, false, false);
#pragma unroll
    for (int r = 0; r < 8; ++r) {
      float a = s0[r] * 0.125f, b = s1[r] * 0.125f;   // 1/sqrt(64)
      float mx = fmaxf(a, b);
#pragma unroll
      for (int m = 8; m >= 1; m >>= 1) mx = fmaxf(mx, __shfl_xor(mx, m, 32));
      float mnew = fmaxf(mrow[r], mx);
      float alpha = __expf(mrow[r] - mnew);
      mrow[r] = mnew;
      float p0 = __expf(a - mnew), p1 = __expf(b - mnew);
      float ps = p0 + p1;
#pragma unroll
      for (int m = 8; m >= 1; m >>= 1) ps += __shfl_xor(ps, m, 32);
      lrow[r] = lrow[r] * alpha + ps;
      o0[r] *= alpha; o1[r] *= alpha; o2[r] *= alpha; o3[r] *= alpha;
      P[w][r + 8 * half][l16]      = (_Float16)p0;   // C layout -> LDS
      P[w][r + 8 * half][16 + l16] = (_Float16)p1;
    }
    const v16h ap = load_frag(&P[w][l16][0], half);  // back as A fragment
    const _Float16* vb = vT + ((size_t)h * HD + l16) * CTX + kb;
    o0 = __builtin_amdgcn_wmma_f32_16x16x32_f16(false, ap, false, load_frag(vb, half), (short)0, o0, false, false);
    o1 = __builtin_amdgcn_wmma_f32_16x16x32_f16(false, ap, false, load_frag(vb + (size_t)16 * CTX, half), (short)0, o1, false, false);
    o2 = __builtin_amdgcn_wmma_f32_16x16x32_f16(false, ap, false, load_frag(vb + (size_t)32 * CTX, half), (short)0, o2, false, false);
    o3 = __builtin_amdgcn_wmma_f32_16x16x32_f16(false, ap, false, load_frag(vb + (size_t)48 * CTX, half), (short)0, o3, false, false);
  }
  _Float16* arow0 = aout + (size_t)(qt * 16) * DIMS + h * HD;
#pragma unroll
  for (int r = 0; r < 8; ++r) {
    float rl = 1.f / lrow[r];
    _Float16* dst = arow0 + (size_t)(r + 8 * half) * DIMS;
    dst[ 0 + l16] = (_Float16)(o0[r] * rl);
    dst[16 + l16] = (_Float16)(o1[r] * rl);
    dst[32 + l16] = (_Float16)(o2[r] * rl);
    dst[48 + l16] = (_Float16)(o3[r] * rl);
  }
}

// ---------------------------------------------------------------------------
extern "C" void kernel_launch(void* const* d_in, const int* in_sizes, int n_in,
                              void* d_out, int out_size, void* d_ws, size_t ws_size,
                              hipStream_t stream) {
  const float* x     = (const float*)d_in[0];
  const float* qn_w  = (const float*)d_in[1];
  const float* qn_b  = (const float*)d_in[2];
  const float* q_w   = (const float*)d_in[3];
  const float* q_b   = (const float*)d_in[4];
  const float* kvn_w = (const float*)d_in[5];
  const float* kvn_b = (const float*)d_in[6];
  const float* kv_w  = (const float*)d_in[7];
  const float* kv_b  = (const float*)d_in[8];
  const float* out_w = (const float*)d_in[9];
  const float* out_b = (const float*)d_in[10];
  const float* ln_w  = (const float*)d_in[11];
  const float* ln_b  = (const float*)d_in[12];
  const float* rh_w  = (const float*)d_in[13];
  const float* rh_b  = (const float*)d_in[14];
  float* out = (float*)d_out;
  (void)in_sizes; (void)n_in; (void)out_size; (void)ws_size;

  char* base = (char*)d_ws;
  size_t off = 0;
  auto take = [&](size_t bytes) -> char* {
    char* p = base + off;
    off = (off + bytes + 255) & ~(size_t)255;
    return p;
  };
  _Float16* lnq16  = (_Float16*)take((size_t)CTX * DIMS * 2);
  _Float16* lnkv16 = (_Float16*)take((size_t)CTX * DIMS * 2);
  float*    radius = (float*)take((size_t)CTX * 4);
  _Float16* qw16   = (_Float16*)take((size_t)DIMS * DIMS * 2);
  _Float16* kvw16  = (_Float16*)take((size_t)DIMS * 2 * DIMS * 2);
  _Float16* outw16 = (_Float16*)take((size_t)DIMS * DIMS * 2);
  float*    qf32   = (float*)take((size_t)CTX * DIMS * 4);
  float*    kvf32  = (float*)take((size_t)CTX * 2 * DIMS * 4);
  _Float16* qn16   = (_Float16*)take((size_t)HEAD * CTX * HD * 2);
  _Float16* kn16   = (_Float16*)take((size_t)HEAD * CTX * HD * 2);
  _Float16* vT16   = (_Float16*)take((size_t)HEAD * HD * CTX * 2);
  _Float16* a16    = (_Float16*)take((size_t)CTX * DIMS * 2);

  prep_kernel<<<CTX, 256, 0, stream>>>(x, qn_w, qn_b, kvn_w, kvn_b, rh_w, rh_b,
                                       lnq16, lnkv16, radius);
  cvt_kernel<<<256, 256, 0, stream>>>(q_w,   qw16,   DIMS * DIMS);
  cvt_kernel<<<256, 256, 0, stream>>>(kv_w,  kvw16,  DIMS * 2 * DIMS);
  cvt_kernel<<<256, 256, 0, stream>>>(out_w, outw16, DIMS * DIMS);

  gemm_kernel<<<dim3(DIMS / BN, CTX / BM), 256, 0, stream>>>(
      lnq16, qw16, q_b, qf32, CTX, DIMS, DIMS);
  gemm_kernel<<<dim3(2 * DIMS / BN, CTX / BM), 256, 0, stream>>>(
      lnkv16, kvw16, kv_b, kvf32, CTX, 2 * DIMS, DIMS);

  rotln_kernel<<<dim3(CTX, HEAD), 32, 0, stream>>>(
      qf32, kvf32, radius, ln_w, ln_b, qn16, kn16, vT16);

  flash_kernel<<<dim3(CTX / 16 / 4, HEAD), 128, 0, stream>>>(qn16, kn16, vT16, a16);

  gemm_kernel<<<dim3(DIMS / BN, CTX / BM), 256, 0, stream>>>(
      a16, outw16, out_b, out, CTX, DIMS, DIMS);
}
